// SpaceTimePolarSSM_87479893885015
// MI455X (gfx1250) — matrix operand order repaired and checked by reading, not compiled, not run on hardware
//
#include <hip/hip_runtime.h>
#include <cstdint>
#include <cstddef>

// ---------------------------------------------------------------------------
// SpaceTimePolarSSM for MI455X (gfx1250, wave32, WMMA + async-to-LDS).
// GEMMs: bf16 operands (pre-converted/transposed), v_wmma_f32_16x16x32_bf16,
// B tile staged into LDS via global_load_async_to_lds_b128 (ASYNCcnt).
// SSM scan is register-resident, one wave per (sequence, head).
// ---------------------------------------------------------------------------

typedef __attribute__((ext_vector_type(16))) __bf16 v16bf;
typedef __attribute__((ext_vector_type(8)))  __bf16 v8bf;
typedef __attribute__((ext_vector_type(8)))  float  v8f;

__device__ __forceinline__ __bf16 f2bf(float f) {
  uint32_t u = __builtin_bit_cast(uint32_t, f);
  uint32_t r = (u + 0x7fffu + ((u >> 16) & 1u)) >> 16;   // round-to-nearest-even
  uint16_t h = (uint16_t)r;
  return __builtin_bit_cast(__bf16, h);
}

__device__ __forceinline__ float blk_sum(float v, float* red, int tid, int n) {
  red[tid] = v; __syncthreads();
  for (int o = n >> 1; o > 0; o >>= 1) {
    if (tid < o) red[tid] += red[tid + o];
    __syncthreads();
  }
  float s = red[0]; __syncthreads();
  return s;
}

__device__ __forceinline__ float siluf(float x) { return x / (1.f + __expf(-x)); }

// ---------------------------------------------------------------------------
// Weight prep: Wt[n*K + k] = bf16(W[k*N + n])   (transpose + convert, once)
// ---------------------------------------------------------------------------
__global__ void prep_weight_kernel(const float* __restrict__ W, __bf16* __restrict__ Wt,
                                   int K, int N) {
  int gid = blockIdx.x * blockDim.x + threadIdx.x;
  if (gid >= K * N) return;
  int k = gid % K, n = gid / K;
  Wt[(size_t)n * K + k] = f2bf(W[(size_t)k * N + n]);
}

// ---------------------------------------------------------------------------
// GEMM: C[M,N] f32 = A[M,K] bf16 (row-major) * Bt[N,K] bf16 (pre-transposed).
// Block = 256 threads (8 waves): 128 rows x 16 cols. K % 32 == 0 (128 / 256).
// Per k-step: B tile (16 cols x 32 k = 1KB) async-staged to LDS by wave 0;
// A fragment = two contiguous 16B global loads per lane (ISA 7.12.2 layout).
// ---------------------------------------------------------------------------
__global__ void gemm_bf16_wmma(const __bf16* __restrict__ A,
                               const __bf16* __restrict__ Bt,
                               float* __restrict__ C,
                               int M, int N, int K) {
  const int wave = threadIdx.x >> 5;
  const int lane = threadIdx.x & 31;
  const int hh   = lane >> 4;        // lane half
  const int lr   = lane & 15;
  const int m0 = blockIdx.x * 128 + wave * 16;
  const int n0 = blockIdx.y * 16;

  __shared__ v8bf stageB[16][4];     // [col-in-tile][4 x 8 bf16] = 32 k values/row

  int arow = m0 + lr; if (arow >= M) arow = M - 1;       // clamp keeps EXEC full
  const __bf16* __restrict__ Arow = A + (size_t)arow * K;

  v8f acc = {};
  for (int k0 = 0; k0 < K; k0 += 32) {
    // ---- stage B tile into LDS (wave 0, ASYNCcnt protocol) ----
    if (wave == 0) {
#pragma unroll
      for (int c = lane; c < 64; c += 32) {        // 16 rows x 4 x 16B chunks
        int row = c >> 2, part = c & 3;
        int nn = n0 + row; if (nn >= N) nn = N - 1;
        const __bf16* gptr = Bt + (size_t)nn * K + k0 + part * 8;
        unsigned lds = (unsigned)(uintptr_t)(void*)&stageB[row][part];
        asm volatile("global_load_async_to_lds_b128 %0, %1, off"
                     :: "v"(lds), "v"(gptr) : "memory");
      }
      asm volatile("s_wait_asynccnt 0" ::: "memory");
    }
    __syncthreads();

    if (k0 + 32 < K) {
      __builtin_prefetch((const void*)(Arow + k0 + 32), 0, 0);   // global_prefetch_b8
    }

    // ---- A fragment: lanes0-15 hold K{0..7,16..23}, lanes16-31 K{8..15,24..31}
    v8bf a0 = *(const v8bf*)(Arow + k0 + 8 * hh);
    v8bf a1 = *(const v8bf*)(Arow + k0 + 16 + 8 * hh);
    v16bf af = __builtin_shufflevector(a0, a1, 0, 1, 2, 3, 4, 5, 6, 7,
                                       8, 9, 10, 11, 12, 13, 14, 15);
    // ---- B fragment from LDS: lane holds col lr, K{8hh..8hh+7, 16+8hh..23+8hh}
    v8bf b0 = stageB[lr][hh];
    v8bf b1 = stageB[lr][2 + hh];
    v16bf bff = __builtin_shufflevector(b0, b1, 0, 1, 2, 3, 4, 5, 6, 7,
                                        8, 9, 10, 11, 12, 13, 14, 15);

    acc = __builtin_amdgcn_wmma_f32_16x16x32_bf16(
        /*neg_a=*/false, af, /*neg_b=*/false, bff,
        /*c_mod=*/(short)0, acc, /*reuse_a=*/false, /*reuse_b=*/false);
    __syncthreads();                 // protect stageB before next overwrite
  }

  // C/D layout: vgpr r -> (M=r, N=lane) lanes0-15 ; (M=8+r, N=lane-16) lanes16-31
  const int n = n0 + lr;
  const int mb = m0 + hh * 8;
#pragma unroll
  for (int r = 0; r < 8; ++r) {
    int m = mb + r;
    if (m < M && n < N) C[(size_t)m * N + n] = acc[r];
  }
}

// ---------------------------------------------------------------------------
// Stage 1: per-token feature extraction (mag / phase) + proj + gelu + LN
// ---------------------------------------------------------------------------
__global__ void feature_kernel(const float* __restrict__ zr, const float* __restrict__ zi,
                               const int*   __restrict__ coords,
                               const float* __restrict__ Wm, const float* __restrict__ bm,
                               const float* __restrict__ gmg, const float* __restrict__ gmb,
                               const float* __restrict__ Wp, const float* __restrict__ bp,
                               const float* __restrict__ gpg, const float* __restrict__ gpb,
                               float* __restrict__ fmag, float* __restrict__ fphase) {
  const int tok = blockIdx.x;
  const int tid = threadIdx.x;          // 128
  __shared__ float sm[66];
  __shared__ float sp[130];
  __shared__ float red[128];

  float zrv = 0.f, ziv = 0.f, m = 0.f;
  if (tid < 64) {
    zrv = zr[(size_t)tok * 64 + tid];
    ziv = zi[(size_t)tok * 64 + tid];
    m = sqrtf(zrv * zrv + ziv * ziv);
  }
  float ss = blk_sum(tid < 64 ? m * m : 0.f, red, tid, 128);
  float nrm = fmaxf(sqrtf(ss), 1e-12f);
  if (tid < 64) {
    sm[tid] = m / nrm;
    sp[tid]      = (m > 0.f) ? zrv / m : 1.f;   // cos(atan2)
    sp[64 + tid] = (m > 0.f) ? ziv / m : 0.f;   // sin(atan2)
  }
  if (tid == 0) {
    float xn = (float)coords[(size_t)tok * 2 + 0] / 346.0f;
    float yn = (float)coords[(size_t)tok * 2 + 1] / 260.0f;
    sm[64] = xn; sm[65] = yn; sp[128] = xn; sp[129] = yn;
  }
  __syncthreads();

  float a = bm[tid];
  for (int i = 0; i < 66; ++i) a += sm[i] * Wm[i * 128 + tid];
  a = 0.5f * a * (1.f + erff(a * 0.70710678118f));
  float mu  = blk_sum(a, red, tid, 128) * (1.f / 128.f);
  float var = blk_sum((a - mu) * (a - mu), red, tid, 128) * (1.f / 128.f);
  fmag[(size_t)tok * 128 + tid] = (a - mu) * rsqrtf(var + 1e-5f) * gmg[tid] + gmb[tid];

  float p = bp[tid];
  for (int i = 0; i < 130; ++i) p += sp[i] * Wp[i * 128 + tid];
  p = 0.5f * p * (1.f + erff(p * 0.70710678118f));
  mu  = blk_sum(p, red, tid, 128) * (1.f / 128.f);
  var = blk_sum((p - mu) * (p - mu), red, tid, 128) * (1.f / 128.f);
  fphase[(size_t)tok * 128 + tid] = (p - mu) * rsqrtf(var + 1e-5f) * gpg[tid] + gpb[tid];
}

// ---------------------------------------------------------------------------
// Bitonic argsort of 512 keys per sequence (stable via (key<<32)|idx packing).
// ---------------------------------------------------------------------------
__global__ void sort_kernel(const int* __restrict__ coords, int order,
                            int* __restrict__ idx, int* __restrict__ inv) {
  const int s = blockIdx.x, tid = threadIdx.x;   // 256
  __shared__ unsigned long long a[512];
  for (int t = tid; t < 512; t += 256) {
    int c0 = coords[((size_t)s * 512 + t) * 2 + 0];
    int c1 = coords[((size_t)s * 512 + t) * 2 + 1];
    unsigned int key = (order == 0) ? (unsigned int)(c1 * 346 + c0)
                                    : (unsigned int)(c0 * 260 + c1);
    a[t] = ((unsigned long long)key << 32) | (unsigned int)t;
  }
  __syncthreads();
  for (int k = 2; k <= 512; k <<= 1) {
    for (int j = k >> 1; j > 0; j >>= 1) {
      for (int t = tid; t < 512; t += 256) {
        int ixj = t ^ j;
        if (ixj > t) {
          unsigned long long x = a[t], y = a[ixj];
          bool up = ((t & k) == 0);
          if (up ? (x > y) : (x < y)) { a[t] = y; a[ixj] = x; }
        }
      }
      __syncthreads();
    }
  }
  for (int t = tid; t < 512; t += 256) {
    int iv = (int)(a[t] & 0xffffffffu);
    idx[(size_t)s * 512 + t] = iv;
    inv[(size_t)s * 512 + iv] = t;
  }
}

// gather (permute) + f32 -> bf16: xs[row] = bf16(x[idx[row]])
__global__ void gather_kernel(const float* __restrict__ x, const int* __restrict__ idx,
                              __bf16* __restrict__ xs, int total) {
  int gid = blockIdx.x * blockDim.x + threadIdx.x;
  if (gid >= total) return;
  int d = gid & 127; int row = gid >> 7; int s = row >> 9;
  xs[(size_t)row * 128 + d] = f2bf(x[((size_t)s * 512 + idx[row]) * 128 + d]);
}

__global__ void scatter_add_kernel(const float* __restrict__ outp, const int* __restrict__ inv,
                                   float* __restrict__ acc, int total) {
  int gid = blockIdx.x * blockDim.x + threadIdx.x;
  if (gid >= total) return;
  int d = gid & 127; int row = gid >> 7; int s = row >> 9;
  acc[(size_t)row * 128 + d] += outp[((size_t)s * 512 + inv[row]) * 128 + d];
}

// ---------------------------------------------------------------------------
// Causal depthwise conv (window 4) + SiLU, in-place on the xBC slice.
// ---------------------------------------------------------------------------
__global__ void conv_kernel(float* __restrict__ zx, const float* __restrict__ cw,
                            const float* __restrict__ cb,
                            int S, int L, int P, int CD, int DI) {
  int gid = blockIdx.x * blockDim.x + threadIdx.x;
  if (gid >= S * CD) return;
  int c = gid % CD, s = gid / CD;
  float w0 = cw[c * 4 + 0], w1 = cw[c * 4 + 1], w2 = cw[c * 4 + 2], w3 = cw[c * 4 + 3];
  float b = cb[c];
  float p0 = 0.f, p1 = 0.f, p2 = 0.f;
  float* base = zx + (size_t)s * L * P + DI + c;
  for (int t = 0; t < L; ++t) {
    float v = base[(size_t)t * P];
    float u = b + p0 * w0 + p1 * w1 + p2 * w2 + v * w3;
    base[(size_t)t * P] = siluf(u);
    p0 = p1; p1 = p2; p2 = v;
  }
}

// dt = softplus(raw + dt_bias);  dA = exp(-exp(A_log)*dt)
__global__ void dt_kernel(const float* __restrict__ zx, const float* __restrict__ dt_bias,
                          const float* __restrict__ A_log,
                          float* __restrict__ dtb, float* __restrict__ dAb,
                          int S, int L, int P, int H) {
  int gid = blockIdx.x * blockDim.x + threadIdx.x;
  if (gid >= S * H) return;
  int h = gid % H, s = gid / H;
  float bias = dt_bias[h];
  float Ax = __expf(A_log[h]);
  for (int t = 0; t < L; ++t) {
    size_t r = (size_t)s * L + t;
    float x = zx[r * P + (P - H) + h] + bias;
    float d = (x > 20.f) ? x : log1pf(__expf(x));
    dtb[r * H + h] = d;
    dAb[r * H + h] = __expf(-Ax * d);
  }
}

// ---------------------------------------------------------------------------
// Selective-scan: one wave per (sequence, head), state in VGPRs (2 rows/lane).
// ---------------------------------------------------------------------------
template <int NS>
__global__ void scan_kernel(const float* __restrict__ zx,
                            const float* __restrict__ dtb, const float* __restrict__ dAb,
                            const float* __restrict__ Dp, float* __restrict__ y,
                            int L, int P, int H, int DI) {
  const int s = blockIdx.x / H, h = blockIdx.x % H;
  const int lane = threadIdx.x;            // 32 threads
  const int p0 = lane * 2, p1 = p0 + 1;
  float h0[NS], h1[NS];
#pragma unroll
  for (int n = 0; n < NS; ++n) { h0[n] = 0.f; h1[n] = 0.f; }
  __shared__ float sB[NS];
  __shared__ float sC[NS];
  const float Dh = Dp[h];
  for (int t = 0; t < L; ++t) {
    size_t r = (size_t)s * L + t;
    const float* row = zx + r * P;
    float dA = dAb[r * H + h];
    float dt = dtb[r * H + h];
    float x0 = row[DI + h * 64 + p0];
    float x1 = row[DI + h * 64 + p1];
    for (int n = lane; n < NS; n += 32) sB[n] = row[2 * DI + n];
    for (int n = lane; n < NS; n += 32) sC[n] = row[2 * DI + NS + n];
    __syncthreads();
    float a0 = dt * x0, a1 = dt * x1, y0 = 0.f, y1 = 0.f;
#pragma unroll
    for (int n = 0; n < NS; ++n) {
      h0[n] = h0[n] * dA + a0 * sB[n]; y0 += h0[n] * sC[n];
      h1[n] = h1[n] * dA + a1 * sB[n]; y1 += h1[n] * sC[n];
    }
    y[r * (size_t)DI + h * 64 + p0] = y0 + Dh * x0;
    y[r * (size_t)DI + h * 64 + p1] = y1 + Dh * x1;
    __syncthreads();
  }
}

// ybf = bf16( rmsnorm(y * silu(z)) * norm_w )
__global__ void gatenorm_kernel(const float* __restrict__ y, const float* __restrict__ zx,
                                const float* __restrict__ nw, __bf16* __restrict__ ybf,
                                int P, int DI) {
  size_t r = blockIdx.x;
  int d = threadIdx.x;                     // 256
  __shared__ float red[256];
  float v = y[r * DI + d] * siluf(zx[r * P + d]);
  float ms = blk_sum(v * v, red, d, 256) * (1.f / (float)DI);
  ybf[r * DI + d] = f2bf(v * rsqrtf(ms + 1e-5f) * nw[d]);
}

// (mean+max)/2 pooling over N tokens, layernorm -> seg[s,128] (bf16 for GEMM A)
__global__ void pool_ln_kernel(const float* __restrict__ acc,
                               const float* __restrict__ g, const float* __restrict__ b,
                               __bf16* __restrict__ seg, int N) {
  const int s = blockIdx.x, d = threadIdx.x;  // 128
  __shared__ float red[128];
  float sum = 0.f, mx = -3.4e38f;
  for (int t = 0; t < N; ++t) {
    float v = 0.5f * acc[((size_t)s * N + t) * 128 + d];   // average of 2 scan orders
    sum += v; mx = fmaxf(mx, v);
  }
  float f = (sum / (float)N + mx) * 0.5f;
  float mu  = blk_sum(f, red, d, 128) * (1.f / 128.f);
  float var = blk_sum((f - mu) * (f - mu), red, d, 128) * (1.f / 128.f);
  seg[(size_t)s * 128 + d] = f2bf((f - mu) * rsqrtf(var + 1e-5f) * g[d] + b[d]);
}

// cross-gating + fuse projection, one block of 128 per (b,t)
__global__ void fuse_kernel(const float* __restrict__ gm, const float* __restrict__ gp,
                            const float* __restrict__ Wp2m, const float* __restrict__ bp2m,
                            const float* __restrict__ Wm2p, const float* __restrict__ bm2p,
                            const float* __restrict__ fw, const float* __restrict__ fb,
                            float* __restrict__ fused) {
  const int r = blockIdx.x, j = threadIdx.x;   // 128
  __shared__ float sgm[128], sgp[128], sa[128], sb[128];
  sgm[j] = gm[(size_t)r * 128 + j];
  sgp[j] = gp[(size_t)r * 128 + j];
  __syncthreads();
  float am = bp2m[j], ap = bm2p[j];
  for (int i = 0; i < 128; ++i) {
    am += sgp[i] * Wp2m[i * 128 + j];
    ap += sgm[i] * Wm2p[i * 128 + j];
  }
  sa[j] = sgm[j] * (1.f / (1.f + __expf(-am)));
  sb[j] = sgp[j] * (1.f / (1.f + __expf(-ap)));
  __syncthreads();
  float f = fb[j];
  for (int i = 0; i < 128; ++i)
    f += sa[i] * fw[i * 128 + j] + sb[i] * fw[(128 + i) * 128 + j];
  fused[(size_t)r * 128 + j] = f;
}

// mean over T then classifier head; one block of 128 per batch element
__global__ void cls_kernel(const float* __restrict__ fused,
                           const float* __restrict__ cw, const float* __restrict__ cb,
                           float* __restrict__ out) {
  const int b = blockIdx.x, j = threadIdx.x;   // 128
  __shared__ float pooled[128];
  float s = 0.f;
  for (int t = 0; t < 16; ++t) s += fused[((size_t)b * 16 + t) * 128 + j];
  pooled[j] = s * (1.f / 16.f);
  __syncthreads();
  if (j < 11) {
    float o = cb[j];
    for (int i = 0; i < 128; ++i) o += pooled[i] * cw[i * 11 + j];
    out[b * 11 + j] = o;
  }
}

// ---------------------------------------------------------------------------
// Host-side driver
// ---------------------------------------------------------------------------
namespace {

struct MambaP {
  const float *A_log, *D, *conv_b, *conv_w, *dt_bias, *in_proj, *norm_w, *out_proj;
  const __bf16 *inT, *outT;    // pre-transposed bf16 weights (workspace)
};

MambaP mk_mamba(void* const* d_in, int base) {
  MambaP m;
  m.A_log    = (const float*)d_in[base + 0];
  m.D        = (const float*)d_in[base + 1];
  m.conv_b   = (const float*)d_in[base + 2];
  m.conv_w   = (const float*)d_in[base + 3];
  m.dt_bias  = (const float*)d_in[base + 4];
  m.in_proj  = (const float*)d_in[base + 5];
  m.norm_w   = (const float*)d_in[base + 6];
  m.out_proj = (const float*)d_in[base + 7];
  m.inT = nullptr; m.outT = nullptr;
  return m;
}

void run_mamba(const __bf16* x, float* out, const MambaP& mp,
               int S, int L, int dstate,
               float* zx, float* dtb, float* dAb, float* ybuf, __bf16* ybf,
               hipStream_t stream) {
  const int DI = 256, H = 4;
  const int P  = 2 * DI + 2 * dstate + H;
  const int CD = DI + 2 * dstate;
  const int M  = S * L;
  dim3 g1((M + 127) / 128, (P + 15) / 16);
  gemm_bf16_wmma<<<g1, 256, 0, stream>>>(x, mp.inT, zx, M, P, 128);
  conv_kernel<<<(S * CD + 255) / 256, 256, 0, stream>>>(zx, mp.conv_w, mp.conv_b, S, L, P, CD, DI);
  dt_kernel<<<(S * H + 63) / 64, 64, 0, stream>>>(zx, mp.dt_bias, mp.A_log, dtb, dAb, S, L, P, H);
  if (dstate == 16)
    scan_kernel<16><<<S * H, 32, 0, stream>>>(zx, dtb, dAb, mp.D, ybuf, L, P, H, DI);
  else
    scan_kernel<64><<<S * H, 32, 0, stream>>>(zx, dtb, dAb, mp.D, ybuf, L, P, H, DI);
  gatenorm_kernel<<<M, 256, 0, stream>>>(ybuf, zx, mp.norm_w, ybf, P, DI);
  dim3 g2((M + 127) / 128, (128 + 15) / 16);
  gemm_bf16_wmma<<<g2, 256, 0, stream>>>(ybf, mp.outT, out, M, 128, 256);
}

} // namespace

extern "C" void kernel_launch(void* const* d_in, const int* in_sizes, int n_in,
                              void* d_out, int out_size, void* d_ws, size_t ws_size,
                              hipStream_t stream) {
  (void)in_sizes; (void)n_in; (void)out_size; (void)ws_size;

  // --- input pointers (JAX alphabetical dict flattening of params) ---
  const float* zr     = (const float*)d_in[0];
  const float* zi     = (const float*)d_in[1];
  const int*   coords = (const int*)  d_in[2];
  const float* cls_b  = (const float*)d_in[3];
  const float* cls_w  = (const float*)d_in[4];
  const float* fuse_b = (const float*)d_in[5];
  const float* fuse_w = (const float*)d_in[6];
  const float* gm2p_b = (const float*)d_in[7];
  const float* gm2p_w = (const float*)d_in[8];
  const float* gp2m_b = (const float*)d_in[9];
  const float* gp2m_w = (const float*)d_in[10];
  MambaP gmag   = mk_mamba(d_in, 11);
  MambaP gphase = mk_mamba(d_in, 19);
  const float* lmag_nb = (const float*)d_in[27];
  const float* lmag_ng = (const float*)d_in[28];
  MambaP lmag   = mk_mamba(d_in, 29);
  const float* lph_nb  = (const float*)d_in[37];
  const float* lph_ng  = (const float*)d_in[38];
  MambaP lph    = mk_mamba(d_in, 39);
  const float* magp_b  = (const float*)d_in[47];
  const float* magp_w  = (const float*)d_in[48];
  const float* nm_b    = (const float*)d_in[49];
  const float* nm_g    = (const float*)d_in[50];
  const float* np_b    = (const float*)d_in[51];
  const float* np_g    = (const float*)d_in[52];
  const float* php_b   = (const float*)d_in[53];
  const float* php_w   = (const float*)d_in[54];
  float* out = (float*)d_out;

  const int BT = 64, N = 512, HID = 128;
  const int PL = 548, PG = 644;     // d_in_proj local / global

  // --- workspace carve (256B aligned) ---
  char* w = (char*)d_ws;
  auto alloc = [&](size_t nbytes) -> void* {
    void* p = (void*)w;
    w += (nbytes + 255) & ~(size_t)255;
    return p;
  };
  float*  fmag   = (float*) alloc((size_t)BT * N * HID * 4);
  float*  fphase = (float*) alloc((size_t)BT * N * HID * 4);
  __bf16* xs     = (__bf16*)alloc((size_t)BT * N * HID * 2);
  float*  zx     = (float*) alloc((size_t)BT * N * PL * 4);
  float*  ybuf   = (float*) alloc((size_t)BT * N * 256 * 4);
  __bf16* ybf    = (__bf16*)alloc((size_t)BT * N * 256 * 2);
  float*  outbuf = (float*) alloc((size_t)BT * N * HID * 4);
  float*  acc    = (float*) alloc((size_t)BT * N * HID * 4);
  float*  dtb    = (float*) alloc((size_t)BT * N * 4 * 4);
  float*  dAb    = (float*) alloc((size_t)BT * N * 4 * 4);
  int*    idx    = (int*)   alloc((size_t)BT * N * 4);
  int*    inv    = (int*)   alloc((size_t)BT * N * 4);
  __bf16* segm   = (__bf16*)alloc((size_t)BT * HID * 2);
  __bf16* segp   = (__bf16*)alloc((size_t)BT * HID * 2);
  float*  gmb    = (float*) alloc((size_t)BT * HID * 4);
  float*  gpb    = (float*) alloc((size_t)BT * HID * 4);
  float*  fusedb = (float*) alloc((size_t)BT * HID * 4);
  // transposed bf16 weights
  __bf16* lmag_inT  = (__bf16*)alloc((size_t)128 * PL * 2);
  __bf16* lmag_outT = (__bf16*)alloc((size_t)256 * HID * 2);
  __bf16* lph_inT   = (__bf16*)alloc((size_t)128 * PL * 2);
  __bf16* lph_outT  = (__bf16*)alloc((size_t)256 * HID * 2);
  __bf16* gmag_inT  = (__bf16*)alloc((size_t)128 * PG * 2);
  __bf16* gmag_outT = (__bf16*)alloc((size_t)256 * HID * 2);
  __bf16* gph_inT   = (__bf16*)alloc((size_t)128 * PG * 2);
  __bf16* gph_outT  = (__bf16*)alloc((size_t)256 * HID * 2);
  lmag.inT = lmag_inT;   lmag.outT = lmag_outT;
  lph.inT  = lph_inT;    lph.outT  = lph_outT;
  gmag.inT = gmag_inT;   gmag.outT = gmag_outT;
  gphase.inT = gph_inT;  gphase.outT = gph_outT;

  // --- stage 0: weight prep (convert + transpose to bf16) ---
  auto prep = [&](const float* W, __bf16* Wt, int K, int Nn) {
    int tot = K * Nn;
    prep_weight_kernel<<<(tot + 255) / 256, 256, 0, stream>>>(W, Wt, K, Nn);
  };
  prep(lmag.in_proj,    (__bf16*)lmag_inT,  128, PL);
  prep(lmag.out_proj,   (__bf16*)lmag_outT, 256, HID);
  prep(lph.in_proj,     (__bf16*)lph_inT,   128, PL);
  prep(lph.out_proj,    (__bf16*)lph_outT,  256, HID);
  prep(gmag.in_proj,    (__bf16*)gmag_inT,  128, PG);
  prep(gmag.out_proj,   (__bf16*)gmag_outT, 256, HID);
  prep(gphase.in_proj,  (__bf16*)gph_inT,   128, PG);
  prep(gphase.out_proj, (__bf16*)gph_outT,  256, HID);

  // --- stage 1: features ---
  feature_kernel<<<BT * N, 128, 0, stream>>>(zr, zi, coords,
      magp_w, magp_b, nm_g, nm_b, php_w, php_b, np_g, np_b, fmag, fphase);

  // --- stage 2: local SSM aggregation (mag then phase, two scan orders each) ---
  const int TOT = BT * N * HID;
  for (int branch = 0; branch < 2; ++branch) {
    const float* feat = branch == 0 ? fmag : fphase;
    const MambaP& mp  = branch == 0 ? lmag : lph;
    const float* ng   = branch == 0 ? lmag_ng : lph_ng;
    const float* nb   = branch == 0 ? lmag_nb : lph_nb;
    __bf16* seg       = branch == 0 ? segm : segp;
    hipMemsetAsync(acc, 0, (size_t)TOT * 4, stream);
    for (int order = 0; order < 2; ++order) {
      sort_kernel<<<BT, 256, 0, stream>>>(coords, order, idx, inv);
      gather_kernel<<<(TOT + 255) / 256, 256, 0, stream>>>(feat, idx, xs, TOT);
      run_mamba(xs, outbuf, mp, BT, N, /*dstate=*/16, zx, dtb, dAb, ybuf, ybf, stream);
      scatter_add_kernel<<<(TOT + 255) / 256, 256, 0, stream>>>(outbuf, inv, acc, TOT);
    }
    pool_ln_kernel<<<BT, 128, 0, stream>>>(acc, ng, nb, seg, N);
  }

  // --- stage 3: global SSM over T=16 (B=4 sequences), d_state=64 ---
  run_mamba(segm, gmb, gmag,   4, 16, /*dstate=*/64, zx, dtb, dAb, ybuf, ybf, stream);
  run_mamba(segp, gpb, gphase, 4, 16, /*dstate=*/64, zx, dtb, dAb, ybuf, ybf, stream);

  // --- stage 4: cross-gating, fusion, pooling, classification ---
  fuse_kernel<<<BT, 128, 0, stream>>>(gmb, gpb, gp2m_w, gp2m_b, gm2p_w, gm2p_b,
                                      fuse_w, fuse_b, fusedb);
  cls_kernel<<<4, 128, 0, stream>>>(fusedb, cls_w, cls_b, out);
}
